// LocalEncoder_36395552866573
// MI455X (gfx1250) — matrix operand order repaired and checked by reading, not compiled, not run on hardware
//
#include <hip/hip_runtime.h>
#include <hip/hip_bf16.h>

typedef __attribute__((ext_vector_type(16))) _Float16 v16h;
typedef __attribute__((ext_vector_type(8)))  _Float16 v8h;
typedef __attribute__((ext_vector_type(8)))  float    v8f;

#define DD 64

// ---------------- WMMA fragment helpers (layouts per CDNA5 ISA 7.12.2) ----

// A fragment: 16x32 f16, row-major source in LDS with row stride `strideA` halves.
// lane 0-15: M=lane, K = {0..7, 16..23}; lane 16-31: M=lane-16, K = {8..15, 24..31}.
__device__ __forceinline__ v16h lda_frag(const _Float16* A, int strideA, int k0, int lane) {
  int m    = lane & 15;
  int koff = (lane >> 4) * 8;
  const _Float16* base = A + m * strideA + k0 + koff;
  v8h lo = *(const v8h*)(base);
  v8h hi = *(const v8h*)(base + 16);
  v16h r;
#pragma unroll
  for (int i = 0; i < 8; ++i) { r[i] = lo[i]; r[i + 8] = hi[i]; }
  return r;
}

// B fragment from pre-swizzled packed weights: tile (kt,ct) of a [K,N] matrix,
// each lane's 16 halfs stored contiguously (32B aligned single load).
__device__ __forceinline__ v16h ldb_frag(const _Float16* P, int nct, int kt, int ct, int lane) {
  return ((const v16h*)P)[((size_t)(kt * nct + ct)) * 32 + lane];
}

__device__ __forceinline__ void zero4(v8f acc[4]) {
#pragma unroll
  for (int c = 0; c < 4; ++c)
#pragma unroll
    for (int r = 0; r < 8; ++r) acc[c][r] = 0.f;
}

// 16 rows x 64 cols GEMM slice: D[16, ctBase*16 .. ctBase*16+63] += A[16,K] * B[K, ...]
__device__ __forceinline__ void gemm4(const _Float16* A, int strideA,
                                      const _Float16* P, int nct, int ctBase,
                                      int kTiles, v8f acc[4], int lane) {
  for (int kt = 0; kt < kTiles; ++kt) {
    v16h a = lda_frag(A, strideA, kt * 32, lane);
#pragma unroll
    for (int c = 0; c < 4; ++c) {
      v16h b = ldb_frag(P, nct, kt, ctBase + c, lane);
      acc[c] = __builtin_amdgcn_wmma_f32_16x16x32_f16(
          false, a, false, b, (short)0, acc[c], false, false);
    }
  }
}

// Store acc (+bias) into f32 LDS tile [16][64]; D-layout: m=r+8*(lane>>4), n=c*16+(lane&15)
__device__ __forceinline__ void acc_to_lds(float* fb, const v8f acc[4],
                                           const float* bias, int lane) {
  int n0 = lane & 15, half = lane >> 4;
#pragma unroll
  for (int c = 0; c < 4; ++c) {
    int n = c * 16 + n0;
    float b = bias ? bias[n] : 0.f;
#pragma unroll
    for (int r = 0; r < 8; ++r) {
      int m = r + 8 * half;
      fb[m * DD + n] = acc[c][r] + b;
    }
  }
}

// Row-wise LayerNorm over 16x64 f32 LDS tile. 2 lanes per row, combine via shfl_xor(16).
__device__ __forceinline__ void ln_rows(float* fb, _Float16* hb, bool relu,
                                        bool writeback_f32, int lane) {
  int row = lane & 15, half = lane >> 4;
  float vals[32];
  float s = 0.f, s2 = 0.f;
#pragma unroll
  for (int j = 0; j < 32; ++j) {
    float v = fb[row * DD + half * 32 + j];
    vals[j] = v; s += v; s2 += v * v;
  }
  s  += __shfl_xor(s, 16);
  s2 += __shfl_xor(s2, 16);
  float mean = s * 0.015625f;
  float var  = s2 * 0.015625f - mean * mean;
  float inv  = rsqrtf(var + 1e-5f);
#pragma unroll
  for (int j = 0; j < 32; ++j) {
    float o = (vals[j] - mean) * inv;
    if (relu) o = fmaxf(o, 0.f);
    if (hb) hb[row * DD + half * 32 + j] = (_Float16)o;
    if (writeback_f32) fb[row * DD + half * 32 + j] = o;
  }
}

__device__ __forceinline__ void atomicMaxFloat(float* addr, float val) {
  if (val >= 0.f) atomicMax((int*)addr, __float_as_int(val));
  else            atomicMin((unsigned int*)addr, __float_as_uint(val));
}

__device__ __forceinline__ float sigmoidf(float x) { return 1.f / (1.f + __expf(-x)); }

// ---------------- Prepack: f32 [K,N] weight -> f16 B-fragment swizzled ------------

__global__ void prepack_kernel(const float* __restrict__ W, _Float16* __restrict__ P, int N) {
  int nct  = N >> 4;
  int tile = blockIdx.x;
  int kt = tile / nct, ct = tile % nct;
  int lane = threadIdx.x;
  int kb = kt * 32 + ((lane >> 4) << 3);
  int n  = ct * 16 + (lane & 15);
  _Float16* dst = P + ((size_t)tile * 32 + lane) * 16;
#pragma unroll
  for (int e = 0; e < 16; ++e) dst[e] = (_Float16)W[(size_t)(kb + e + ((e >> 3) << 3)) * N + n];
}

// ---------------- K0: init reduction buffers ----------------

__global__ void k0_init(float* amax, float* den, float* agg, int N) {
  int i = blockIdx.x * blockDim.x + threadIdx.x;
  if (i < N * DD) agg[i] = 0.f;
  if (i < N * 8) { amax[i] = __uint_as_float(0xff800000u); den[i] = 0.f; }
}

// ---------------- K1: node SIE -> center, c1, q  (8 waves, 16 nodes each) ---------

__global__ __launch_bounds__(256) void k1_nodes(
    const float* __restrict__ x, const float* __restrict__ rot,
    const unsigned char* __restrict__ bos_mask, const int* __restrict__ tptr,
    const float* __restrict__ bos_token,
    const float* __restrict__ ce_w1, const float* __restrict__ ce_b1,
    const float* __restrict__ ce_b2, const float* __restrict__ ce_b3,
    const float* __restrict__ q_b,
    const _Float16* __restrict__ P_ce2, const _Float16* __restrict__ P_ce3,
    const _Float16* __restrict__ P_q,
    float* __restrict__ center, float* __restrict__ c1, float* __restrict__ q, int N) {
  __shared__ __align__(32) float    fbuf[8][16 * DD];
  __shared__ __align__(32) _Float16 hbuf[8][16 * DD];
  __shared__ float xt[8][16][2];
  int wave = threadIdx.x >> 5, lane = threadIdx.x & 31;
  int tile = blockIdx.x * 8 + wave;
  int nb = tile * 16;
  int row = lane & 15, half = lane >> 4;
  int nid = nb + row; if (nid > N - 1) nid = N - 1;
  bool rv = (nb + row) < N;
  float* fb = fbuf[wave]; _Float16* hb = hbuf[wave];

  if (half == 0) {
    float x0 = x[(size_t)nid * 2], x1 = x[(size_t)nid * 2 + 1];
    float r00 = rot[(size_t)nid * 4], r01 = rot[(size_t)nid * 4 + 1];
    float r10 = rot[(size_t)nid * 4 + 2], r11 = rot[(size_t)nid * 4 + 3];
    xt[wave][row][0] = x0 * r00 + x1 * r10;
    xt[wave][row][1] = x0 * r01 + x1 * r11;
  }
  __syncthreads();
  {
    float a0 = xt[wave][row][0], a1 = xt[wave][row][1];
#pragma unroll
    for (int j = 0; j < 32; ++j) {
      int col = half * 32 + j;
      fb[row * DD + col] = a0 * ce_w1[col] + a1 * ce_w1[DD + col] + ce_b1[col];
    }
  }
  __syncthreads();
  ln_rows(fb, hb, true, false, lane);
  __syncthreads();
  v8f acc[4];
  zero4(acc); gemm4(hb, DD, P_ce2, 4, 0, 2, acc, lane);
  __syncthreads();
  acc_to_lds(fb, acc, ce_b2, lane);
  __syncthreads();
  ln_rows(fb, hb, true, false, lane);
  __syncthreads();
  zero4(acc); gemm4(hb, DD, P_ce3, 4, 0, 2, acc, lane);
  __syncthreads();
  acc_to_lds(fb, acc, ce_b3, lane);
  __syncthreads();
  ln_rows(fb, (_Float16*)0, false, true, lane);  // fb = SIE output (center pre-bos)
  __syncthreads();
  {
    int t = tptr[0];
    bool bos = bos_mask[nid] != 0;
#pragma unroll
    for (int j = 0; j < 32; ++j) {
      int col = half * 32 + j;
      float cv = fb[row * DD + col];
      if (bos) cv = bos_token[(size_t)t * DD + col];
      fb[row * DD + col] = cv;
      if (rv) center[(size_t)(nb + row) * DD + col] = cv;
    }
  }
  __syncthreads();
  ln_rows(fb, hb, false, true, lane);  // fb,hb = c1
  __syncthreads();
  if (rv) {
#pragma unroll
    for (int j = 0; j < 32; ++j) {
      int col = half * 32 + j;
      c1[(size_t)(nb + row) * DD + col] = fb[row * DD + col];
    }
  }
  zero4(acc); gemm4(hb, DD, P_q, 4, 0, 2, acc, lane);
  __syncthreads();
  acc_to_lds(fb, acc, q_b, lane);
  __syncthreads();
  if (rv) {
#pragma unroll
    for (int j = 0; j < 32; ++j) {
      int col = half * 32 + j;
      q[(size_t)(nb + row) * DD + col] = fb[row * DD + col];
    }
  }
}

// ---------------- K2: edge MIE + k/v + alpha (4 waves, 16 edges each) -------------

__global__ __launch_bounds__(128) void k2_edges(
    const float* __restrict__ x, const float* __restrict__ edge_attr,
    const float* __restrict__ rot, const int* __restrict__ edge_index,
    const float* __restrict__ m0_w1, const float* __restrict__ m0_b1,
    const float* __restrict__ m0_b2,
    const float* __restrict__ m1_w1, const float* __restrict__ m1_b1,
    const float* __restrict__ m1_b2,
    const float* __restrict__ ag_b, const float* __restrict__ k_b,
    const float* __restrict__ v_b,
    const _Float16* __restrict__ P_m0, const _Float16* __restrict__ P_m1,
    const _Float16* __restrict__ P_ag, const _Float16* __restrict__ P_k,
    const _Float16* __restrict__ P_v,
    const float* __restrict__ q,
    float* __restrict__ alpha, float* __restrict__ amax, float* __restrict__ vbuf,
    int N, int E) {
  __shared__ __align__(32) float    fbuf[4][16 * DD];
  __shared__ __align__(32) float    sbuf[4][16 * DD];
  __shared__ __align__(32) _Float16 hbuf[4][16 * DD];
  __shared__ float inbuf[4][16][4];
  __shared__ int   dstbuf[4][16];
  int wave = threadIdx.x >> 5, lane = threadIdx.x & 31;
  int tile = blockIdx.x * 4 + wave;
  int eb = tile * 16;
  int row = lane & 15, half = lane >> 4, n0 = lane & 15;
  int e = eb + row; if (e > E - 1) e = E - 1;
  bool rv = (eb + row) < E;
  float* fb = fbuf[wave]; float* sb = sbuf[wave]; _Float16* hb = hbuf[wave];

  if (half == 0) {
    int s = edge_index[e];
    int d = edge_index[(size_t)E + e];
    dstbuf[wave][row] = d;
    float xs0 = x[(size_t)s * 2], xs1 = x[(size_t)s * 2 + 1];
    float r00 = rot[(size_t)d * 4], r01 = rot[(size_t)d * 4 + 1];
    float r10 = rot[(size_t)d * 4 + 2], r11 = rot[(size_t)d * 4 + 3];
    inbuf[wave][row][0] = xs0 * r00 + xs1 * r10;
    inbuf[wave][row][1] = xs0 * r01 + xs1 * r11;
    float a0 = edge_attr[(size_t)e * 2], a1 = edge_attr[(size_t)e * 2 + 1];
    inbuf[wave][row][2] = a0 * r00 + a1 * r10;
    inbuf[wave][row][3] = a0 * r01 + a1 * r11;
  }
  __syncthreads();
  v8f acc[4];
  // ---- input 0 branch: 2->64, LN, ReLU, @m0_w2 -> sb
  {
    float a0 = inbuf[wave][row][0], a1 = inbuf[wave][row][1];
#pragma unroll
    for (int j = 0; j < 32; ++j) {
      int col = half * 32 + j;
      fb[row * DD + col] = a0 * m0_w1[col] + a1 * m0_w1[DD + col] + m0_b1[col];
    }
  }
  __syncthreads();
  ln_rows(fb, hb, true, false, lane);
  __syncthreads();
  zero4(acc); gemm4(hb, DD, P_m0, 4, 0, 2, acc, lane);
  __syncthreads();
  acc_to_lds(sb, acc, m0_b2, lane);  // e0 in sb
  // ---- input 1 branch
  {
    float a0 = inbuf[wave][row][2], a1 = inbuf[wave][row][3];
#pragma unroll
    for (int j = 0; j < 32; ++j) {
      int col = half * 32 + j;
      fb[row * DD + col] = a0 * m1_w1[col] + a1 * m1_w1[DD + col] + m1_b1[col];
    }
  }
  __syncthreads();
  ln_rows(fb, hb, true, false, lane);
  __syncthreads();
  zero4(acc); gemm4(hb, DD, P_m1, 4, 0, 2, acc, lane);
  {  // sb += e1 (same lane wrote same positions earlier)
#pragma unroll
    for (int c = 0; c < 4; ++c) {
      int n = c * 16 + n0;
#pragma unroll
      for (int r = 0; r < 8; ++r) {
        int m = r + 8 * half;
        sb[m * DD + n] += acc[c][r] + m1_b2[n];
      }
    }
  }
  __syncthreads();
  ln_rows(sb, hb, true, false, lane);
  __syncthreads();
  zero4(acc); gemm4(hb, DD, P_ag, 4, 0, 2, acc, lane);
  __syncthreads();
  acc_to_lds(fb, acc, ag_b, lane);
  __syncthreads();
  ln_rows(fb, hb, false, false, lane);  // hb = nbr (f16)
  __syncthreads();
  // ---- k projection + alpha
  zero4(acc); gemm4(hb, DD, P_k, 4, 0, 2, acc, lane);
  __syncthreads();
  acc_to_lds(fb, acc, k_b, lane);
  __syncthreads();
  {
    int d = dstbuf[wave][row];
#pragma unroll
    for (int hh = 0; hh < 4; ++hh) {
      int h = half * 4 + hh;
      float sdot = 0.f;
#pragma unroll
      for (int t2 = 0; t2 < 8; ++t2)
        sdot += fb[row * DD + h * 8 + t2] * q[(size_t)d * DD + h * 8 + t2];
      sdot *= 0.35355339059327373f;  // 1/sqrt(8)
      if (rv) {
        alpha[(size_t)(eb + row) * 8 + h] = sdot;
        atomicMaxFloat(&amax[(size_t)d * 8 + h], sdot);
      }
    }
  }
  // ---- v projection -> vbuf
  zero4(acc); gemm4(hb, DD, P_v, 4, 0, 2, acc, lane);
  __syncthreads();
  acc_to_lds(fb, acc, v_b, lane);
  __syncthreads();
  if (rv) {
    float4* dst = (float4*)&vbuf[(size_t)(eb + row) * DD + half * 32];
    const float4* src = (const float4*)&fb[row * DD + half * 32];
#pragma unroll
    for (int j = 0; j < 8; ++j) dst[j] = src[j];
  }
}

// ---------------- K3: ex = exp(alpha - amax[dst]); den += ex ----------------------

__global__ void k3_exden(const int* __restrict__ edge_index,
                         const float* __restrict__ amax, float* __restrict__ alpha,
                         float* __restrict__ den, int E) {
  int e = blockIdx.x * blockDim.x + threadIdx.x;
  if (e >= E) return;
  int d = edge_index[(size_t)E + e];
#pragma unroll
  for (int h = 0; h < 8; ++h) {
    float al = alpha[(size_t)e * 8 + h];
    float ex = __expf(al - amax[(size_t)d * 8 + h]);
    alpha[(size_t)e * 8 + h] = ex;
    atomicAdd(&den[(size_t)d * 8 + h], ex);
  }
}

// ---------------- K4: agg[dst] += v * a ----------------------

__global__ void k4_agg(const int* __restrict__ edge_index,
                       const float* __restrict__ exw, const float* __restrict__ den,
                       const float* __restrict__ vbuf, float* __restrict__ agg, int E) {
  int tid = blockIdx.x * blockDim.x + threadIdx.x;
  int e = tid >> 3, h = tid & 7;
  if (e >= E) return;
  int d = edge_index[(size_t)E + e];
  float a = exw[(size_t)e * 8 + h] / (den[(size_t)d * 8 + h] + 1e-16f);
#pragma unroll
  for (int j = 0; j < 8; ++j)
    atomicAdd(&agg[(size_t)d * DD + h * 8 + j], vbuf[(size_t)e * DD + h * 8 + j] * a);
}

// ---------------- K5: gated update + FFN (2 waves, 16 nodes each) -----------------

__global__ __launch_bounds__(64) void k5_update(
    const float* __restrict__ agg, const float* __restrict__ c1,
    const float* __restrict__ center,
    const float* __restrict__ ih_b, const float* __restrict__ hh_b,
    const float* __restrict__ self_b, const float* __restrict__ out_b,
    const float* __restrict__ fc1_b, const float* __restrict__ fc2_b,
    const _Float16* __restrict__ P_ih, const _Float16* __restrict__ P_hh,
    const _Float16* __restrict__ P_self, const _Float16* __restrict__ P_out,
    const _Float16* __restrict__ P_fc1, const _Float16* __restrict__ P_fc2,
    float* __restrict__ outp, int N) {
  __shared__ __align__(32) float    abuf[2][16 * DD];
  __shared__ __align__(32) float    fbuf[2][16 * DD];
  __shared__ __align__(32) _Float16 hA[2][16 * DD];
  __shared__ __align__(32) _Float16 hC[2][16 * DD];
  __shared__ __align__(32) _Float16 h256[2][16 * 256];
  int wave = threadIdx.x >> 5, lane = threadIdx.x & 31;
  int tile = blockIdx.x * 2 + wave;
  int nb = tile * 16;
  int row = lane & 15, half = lane >> 4, n0 = lane & 15;
  int nid = nb + row; if (nid > N - 1) nid = N - 1;
  float* ab = abuf[wave]; float* fb = fbuf[wave];
  _Float16* ha = hA[wave]; _Float16* hc = hC[wave]; _Float16* h2 = h256[wave];

  // load agg (f32+f16) and c1 (f16)
#pragma unroll
  for (int j = 0; j < 8; ++j) {
    float4 v4 = ((const float4*)&agg[(size_t)nid * DD + half * 32])[j];
    ((float4*)&ab[row * DD + half * 32])[j] = v4;
    int base = row * DD + half * 32 + j * 4;
    ha[base + 0] = (_Float16)v4.x; ha[base + 1] = (_Float16)v4.y;
    ha[base + 2] = (_Float16)v4.z; ha[base + 3] = (_Float16)v4.w;
    float4 c4 = ((const float4*)&c1[(size_t)nid * DD + half * 32])[j];
    hc[base + 0] = (_Float16)c4.x; hc[base + 1] = (_Float16)c4.y;
    hc[base + 2] = (_Float16)c4.z; hc[base + 3] = (_Float16)c4.w;
  }
  __syncthreads();
  v8f accG[4], accS[4];
  zero4(accG);
  gemm4(ha, DD, P_ih, 4, 0, 2, accG, lane);
  gemm4(hc, DD, P_hh, 4, 0, 2, accG, lane);  // accumulate
  zero4(accS);
  gemm4(hc, DD, P_self, 4, 0, 2, accS, lane);
  __syncthreads();
  {  // gate + upd -> ha (f16)
#pragma unroll
    for (int c = 0; c < 4; ++c) {
      int n = c * 16 + n0;
#pragma unroll
      for (int r = 0; r < 8; ++r) {
        int m = r + 8 * half;
        float g = sigmoidf(accG[c][r] + ih_b[n] + hh_b[n]);
        float sv = accS[c][r] + self_b[n];
        float ag = ab[m * DD + n];
        float upd = ag + g * (sv - ag);
        ha[m * DD + n] = (_Float16)upd;
      }
    }
  }
  __syncthreads();
  v8f accO[4];
  zero4(accO);
  gemm4(ha, DD, P_out, 4, 0, 2, accO, lane);
  {  // center2 = center + upd@out_w + out_b -> fb
#pragma unroll
    for (int c = 0; c < 4; ++c) {
      int n = c * 16 + n0;
#pragma unroll
      for (int r = 0; r < 8; ++r) {
        int m = r + 8 * half;
        int nr = nb + m; if (nr > N - 1) nr = N - 1;
        fb[m * DD + n] = center[(size_t)nr * DD + n] + accO[c][r] + out_b[n];
      }
    }
  }
  __syncthreads();
  ln_rows(fb, hc, false, false, lane);  // hc = LN(center2); fb keeps center2
  __syncthreads();
  // FFN fc1: 64 -> 256, ReLU, f16 into h2
  for (int chunk = 0; chunk < 4; ++chunk) {
    v8f accF[4];
    zero4(accF);
    gemm4(hc, DD, P_fc1, 16, chunk * 4, 2, accF, lane);
#pragma unroll
    for (int c = 0; c < 4; ++c) {
      int col = (chunk * 4 + c) * 16 + n0;
#pragma unroll
      for (int r = 0; r < 8; ++r) {
        int m = r + 8 * half;
        float val = accF[c][r] + fc1_b[col];
        h2[m * 256 + col] = (_Float16)fmaxf(val, 0.f);
      }
    }
  }
  __syncthreads();
  // fc2: 256 -> 64; out = center2 + ff
  zero4(accO);
  gemm4(h2, 256, P_fc2, 4, 0, 8, accO, lane);
#pragma unroll
  for (int c = 0; c < 4; ++c) {
    int n = c * 16 + n0;
#pragma unroll
    for (int r = 0; r < 8; ++r) {
      int m = r + 8 * half;
      if (nb + m < N)
        outp[(size_t)(nb + m) * DD + n] = fb[m * DD + n] + accO[c][r] + fc2_b[n];
    }
  }
}

// ---------------- host launcher ----------------

extern "C" void kernel_launch(void* const* d_in, const int* in_sizes, int n_in,
                              void* d_out, int out_size, void* d_ws, size_t ws_size,
                              hipStream_t stream) {
  const float* x          = (const float*)d_in[0];
  const int*   tptr       = (const int*)d_in[1];
  const int*   edge_index = (const int*)d_in[2];
  const float* edge_attr  = (const float*)d_in[3];
  const unsigned char* bos_mask = (const unsigned char*)d_in[4];
  const float* rot        = (const float*)d_in[5];
  const float* const* P   = (const float* const*)(d_in + 6);
  // params order: 0 ce_w1,1 ce_b1,2 ce_w2,3 ce_b2,4 ce_w3,5 ce_b3,6 m0_w1,7 m0_b1,
  // 8 m0_w2,9 m0_b2,10 m1_w1,11 m1_b1,12 m1_w2,13 m1_b2,14 ag_w,15 ag_b,16 q_w,17 q_b,
  // 18 k_w,19 k_b,20 v_w,21 v_b,22 self_w,23 self_b,24 ih_w,25 ih_b,26 hh_w,27 hh_b,
  // 28 out_w,29 out_b,30 fc1_w,31 fc1_b,32 fc2_w,33 fc2_b,34 bos_token
  int N = in_sizes[0] / 2;
  int E = in_sizes[2] / 2;

  char* ws = (char*)d_ws;
  size_t off = 0;
  auto take = [&](size_t bytes) -> void* {
    void* p = ws + off;
    off += (bytes + 255) & ~(size_t)255;
    return p;
  };
  float* center = (float*)take((size_t)N * DD * 4);
  float* c1     = (float*)take((size_t)N * DD * 4);
  float* qbuf   = (float*)take((size_t)N * DD * 4);
  float* amax   = (float*)take((size_t)N * 8 * 4);
  float* den    = (float*)take((size_t)N * 8 * 4);
  float* agg    = (float*)take((size_t)N * DD * 4);
  float* alpha  = (float*)take((size_t)E * 8 * 4);
  float* vbuf   = (float*)take((size_t)E * DD * 4);

  // packed weights (f16)
  const int sqIdx[12] = {2, 4, 16, 8, 12, 14, 18, 20, 22, 24, 26, 28};
  _Float16* Pk[12];
  for (int i = 0; i < 12; ++i) Pk[i] = (_Float16*)take((size_t)64 * 64 * 2);
  _Float16* P_fc1 = (_Float16*)take((size_t)64 * 256 * 2);
  _Float16* P_fc2 = (_Float16*)take((size_t)256 * 64 * 2);

  for (int i = 0; i < 12; ++i)
    prepack_kernel<<<(64 / 32) * (64 / 16), 32, 0, stream>>>(P[sqIdx[i]], Pk[i], 64);
  prepack_kernel<<<(64 / 32) * (256 / 16), 32, 0, stream>>>(P[30], P_fc1, 256);
  prepack_kernel<<<(256 / 32) * (64 / 16), 32, 0, stream>>>(P[32], P_fc2, 64);

  // Pk index map: 0 ce_w2,1 ce_w3,2 q_w,3 m0_w2,4 m1_w2,5 ag_w,6 k_w,7 v_w,
  //               8 self_w,9 ih_w,10 hh_w,11 out_w
  {
    int threads = 256;
    int grid = (N * DD + threads - 1) / threads;
    k0_init<<<grid, threads, 0, stream>>>(amax, den, agg, N);
  }
  {
    int tiles = (N + 15) / 16;
    int grid = (tiles + 7) / 8;
    k1_nodes<<<grid, 256, 0, stream>>>(x, rot, bos_mask, tptr, P[34],
                                       P[0], P[1], P[3], P[5], P[17],
                                       Pk[0], Pk[1], Pk[2],
                                       center, c1, qbuf, N);
  }
  {
    int tiles = (E + 15) / 16;
    int grid = (tiles + 3) / 4;
    k2_edges<<<grid, 128, 0, stream>>>(x, edge_attr, rot, edge_index,
                                       P[6], P[7], P[9],
                                       P[10], P[11], P[13],
                                       P[15], P[19], P[21],
                                       Pk[3], Pk[4], Pk[5], Pk[6], Pk[7],
                                       qbuf, alpha, amax, vbuf, N, E);
  }
  {
    int threads = 256;
    int grid = (E + threads - 1) / threads;
    k3_exden<<<grid, threads, 0, stream>>>(edge_index, amax, alpha, den, E);
  }
  {
    int threads = 256;
    long long total = (long long)E * 8;
    int grid = (int)((total + threads - 1) / threads);
    k4_agg<<<grid, threads, 0, stream>>>(edge_index, alpha, den, vbuf, agg, E);
  }
  {
    int tiles = (N + 15) / 16;
    int grid = (tiles + 1) / 2;
    k5_update<<<grid, 64, 0, stream>>>(agg, c1, center,
                                       P[25], P[27], P[23], P[29], P[31], P[33],
                                       Pk[9], Pk[10], Pk[8], Pk[11], P_fc1, P_fc2,
                                       (float*)d_out, N);
  }
}
